// MultiHeadAttention_76725295776375
// MI455X (gfx1250) — matrix operand order repaired and checked by reading, not compiled, not run on hardware
//
#include <hip/hip_runtime.h>
#include <hip/hip_bf16.h>

// ---------------- problem constants ----------------
#define D_MODEL   1024
#define NUM_HEADS 16
#define D_HEAD    64
#define BATCH     2
#define SEQ       2048
#define NROWS     (BATCH * SEQ)      // 4096 token rows

// ---------------- CDNA5 WMMA types ----------------
typedef __bf16        v16bf  __attribute__((ext_vector_type(16)));
typedef float         v8f    __attribute__((ext_vector_type(8)));
typedef unsigned int  u32x4  __attribute__((ext_vector_type(4)));

union Frag { u32x4 u[2]; v16bf v; };   // 32 bytes = 16 bf16 = one WMMA A/B operand

__device__ __forceinline__ unsigned short f2bf(float f) {
  // round-to-nearest-even f32 -> bf16
  unsigned u = __float_as_uint(f);
  u += 0x7FFFu + ((u >> 16) & 1u);
  return (unsigned short)(u >> 16);
}

__device__ __forceinline__ v8f wmma_bf16(const Frag& a, const Frag& b, v8f c) {
  return __builtin_amdgcn_wmma_f32_16x16x32_bf16(false, a.v, false, b.v,
                                                 (short)0, c, false, false);
}

// gfx1250 async memory->LDS copy (128b per lane), tracked by ASYNCcnt.
// lds_off = byte offset of destination within the workgroup LDS allocation
// (low 32 bits of a generic pointer to a __shared__ object), gaddr = global addr.
__device__ __forceinline__ void async_copy_b128(unsigned lds_off,
                                                unsigned long long gaddr) {
  asm volatile("global_load_async_to_lds_b128 %0, %1, off"
               :: "v"(lds_off), "v"(gaddr) : "memory");
}
__device__ __forceinline__ void wait_async0() {
  asm volatile("s_wait_asynccnt 0x0" ::: "memory");
}

// ---------------- kernel 1: f32 -> bf16 (8 elements / thread) ----------------
__global__ void mha_cvt_bf16(const float* __restrict__ in,
                             unsigned short* __restrict__ out, int n8) {
  int i = blockIdx.x * blockDim.x + threadIdx.x;
  if (i >= n8) return;
  const float4* p = (const float4*)in + (size_t)i * 2;
  float4 f0 = p[0], f1 = p[1];
  u32x4 v;
  v.x = f2bf(f0.x) | ((unsigned)f2bf(f0.y) << 16);
  v.y = f2bf(f0.z) | ((unsigned)f2bf(f0.w) << 16);
  v.z = f2bf(f1.x) | ((unsigned)f2bf(f1.y) << 16);
  v.w = f2bf(f1.z) | ((unsigned)f2bf(f1.w) << 16);
  ((u32x4*)out)[i] = v;
}

// ---------------- kernel 2: QKV projection GEMM ----------------
// 32(M) x 64(N) register tile per wave: 2 A-frags + 4 B-frags -> 8 WMMAs / K-step.
// A : x_bf [NROWS, D_MODEL] row-major.  W : [out, in] row-major (y = x @ W^T, so
// B(k,n) = W[n][k] is contiguous in k per lane -> straight b128 loads).
// grid: (NROWS/32, D_MODEL/256), block 128 (4 waves split N).
__global__ void mha_gemm_qkv(const unsigned short* __restrict__ A,
                             const unsigned short* __restrict__ W,
                             unsigned short* __restrict__ dst, int vtrans) {
  const int wave = threadIdx.x >> 5;
  const int lane = threadIdx.x & 31;
  const int half = lane >> 4;
  const int ln   = lane & 15;

  const int m0 = blockIdx.x * 32;                 // 2 m-tiles
  const int n0 = blockIdx.y * 256 + wave * 64;    // 4 n-tiles

  v8f acc[2][4];
#pragma unroll
  for (int mt = 0; mt < 2; ++mt)
#pragma unroll
    for (int nt = 0; nt < 4; ++nt) acc[mt][nt] = {};

  const unsigned short* aptr[2];
  aptr[0] = A + (size_t)(m0 + ln) * D_MODEL;
  aptr[1] = A + (size_t)(m0 + 16 + ln) * D_MODEL;
  const unsigned short* bptr[4];
#pragma unroll
  for (int nt = 0; nt < 4; ++nt)
    bptr[nt] = W + (size_t)(n0 + nt * 16 + ln) * D_MODEL;

  for (int k0 = 0; k0 < D_MODEL; k0 += 32) {
    Frag a[2], b[4];
#pragma unroll
    for (int mt = 0; mt < 2; ++mt) {
      // A 16-bit 16x32: half-lane holds K = half*16 + [0..15] contiguous
      const u32x4* ap = (const u32x4*)(aptr[mt] + k0 + half * 16);
      a[mt].u[0] = ap[0];
      a[mt].u[1] = ap[1];
    }
#pragma unroll
    for (int nt = 0; nt < 4; ++nt) {
      // B 16-bit 32x16: half-lane holds K = half*16 + [0..15] contiguous
      const u32x4* bp = (const u32x4*)(bptr[nt] + k0 + half * 16);
      b[nt].u[0] = bp[0];
      b[nt].u[1] = bp[1];
    }
    if (k0 + 32 < D_MODEL) {
      __builtin_prefetch(aptr[0] + k0 + 32, 0, 3);
      __builtin_prefetch(aptr[1] + k0 + 32, 0, 3);
      __builtin_prefetch(bptr[0] + k0 + 32, 0, 3);
      __builtin_prefetch(bptr[2] + k0 + 32, 0, 3);
    }
#pragma unroll
    for (int nt = 0; nt < 4; ++nt)
#pragma unroll
      for (int mt = 0; mt < 2; ++mt)
        acc[mt][nt] = wmma_bf16(a[mt], b[nt], acc[mt][nt]);
  }

  // D layout: VGPR r -> (M = r + 8*half, N = ln)
#pragma unroll
  for (int mt = 0; mt < 2; ++mt)
#pragma unroll
    for (int nt = 0; nt < 4; ++nt)
#pragma unroll
      for (int r = 0; r < 8; ++r) {
        const int row = m0 + mt * 16 + r + 8 * half;   // b*SEQ + s
        const int col = n0 + nt * 16 + ln;             // h*Dh + d
        const int bb = row >> 11, s = row & (SEQ - 1);
        const int hh = col >> 6,  d = col & 63;
        const unsigned short bits = f2bf(acc[mt][nt][r]);
        if (vtrans)
          dst[(((size_t)(bb * NUM_HEADS + hh)) * D_HEAD + d) * SEQ + s] = bits;
        else
          dst[(((size_t)(bb * NUM_HEADS + hh)) * SEQ + s) * D_HEAD + d] = bits;
      }
}

// ---------------- kernel 3: streaming (flash-style) attention ----------------
// q,k : [B,H,S,Dh] bf16 ; vT : [B,H,Dh,S] bf16 ; o : [NROWS, D_MODEL] bf16
// Block = 4 waves = 64 queries on one (b,h). K/V chunks (32 keys) are staged
// once per block into double-buffered LDS with GLOBAL_LOAD_ASYNC_TO_LDS_B128,
// prefetching chunk c+1 during chunk c's WMMAs; only s_wait_asynccnt is exposed.
// grid: (SEQ/64, B*H), block 128.
__global__ void mha_attn(const unsigned short* __restrict__ q,
                         const unsigned short* __restrict__ k,
                         const unsigned short* __restrict__ vT,
                         unsigned short* __restrict__ o) {
  const int wave = threadIdx.x >> 5;
  const int lane = threadIdx.x & 31;
  const int half = lane >> 4;
  const int ln   = lane & 15;
  const int tid  = threadIdx.x;

  const int q0 = blockIdx.x * 64 + wave * 16;   // this wave's query tile
  const int bh = blockIdx.y;                    // 0..31
  const int b  = bh / NUM_HEADS;
  const int h  = bh % NUM_HEADS;

  __shared__ unsigned short kbuf[2][32][64];    // [buf][key][d]   8 KB
  __shared__ unsigned short vbuf[2][64][32];    // [buf][d][key]   8 KB
  __shared__ unsigned short pbuf[4][16][32];    // per-wave P relayout 4 KB

  const unsigned klds0 = (unsigned)(size_t)&kbuf[0][0][0];
  const unsigned vlds0 = (unsigned)(size_t)&vbuf[0][0][0];

  // ---- async staging of one 32-key chunk into buffer p ----
  auto stage = [&](int c) {
    const int p = (c >> 5) & 1;
    // K chunk: 32 consecutive [S,Dh] rows => contiguous 4 KB block
    {
      const unsigned long long kg =
          (unsigned long long)(k + ((size_t)bh * SEQ + c) * D_HEAD);
      const unsigned kl = klds0 + (unsigned)p * 4096u + (unsigned)tid * 16u;
      async_copy_b128(kl,           kg + (unsigned long long)tid * 16u);
      async_copy_b128(kl + 2048u,   kg + (unsigned long long)tid * 16u + 2048u);
    }
    // V chunk: 64 vT rows x 32 keys (row stride SEQ elements)
#pragma unroll
    for (int j = 0; j < 2; ++j) {
      const int i    = tid * 2 + j;             // 0..255 b128 tiles
      const int d    = i >> 2;
      const int part = i & 3;
      const unsigned long long vg =
          (unsigned long long)(vT + ((size_t)bh * D_HEAD + d) * SEQ + c) +
          (unsigned long long)part * 16u;
      const unsigned vl =
          vlds0 + (unsigned)p * 4096u + (unsigned)d * 64u + (unsigned)part * 16u;
      async_copy_b128(vl, vg);
    }
  };

  // Q A-fragments for Dh K-steps 0 and 32 (register-resident for whole sweep)
  Frag aq0, aq1;
  {
    const unsigned short* qp = q + ((size_t)bh * SEQ + q0 + ln) * D_HEAD;
    const u32x4* a0 = (const u32x4*)(qp + half * 16);
    aq0.u[0] = a0[0]; aq0.u[1] = a0[1];
    const u32x4* a1 = (const u32x4*)(qp + 32 + half * 16);
    aq1.u[0] = a1[0]; aq1.u[1] = a1[1];
  }

  float m[8], l[8];
  v8f oacc[4];
#pragma unroll
  for (int r = 0; r < 8; ++r) { m[r] = -3.0e38f; l[r] = 0.0f; }
#pragma unroll
  for (int t = 0; t < 4; ++t) oacc[t] = {};

  stage(0);                                     // prime the pipeline

  for (int c = 0; c < SEQ; c += 32) {           // 32-key chunks
    const int p = (c >> 5) & 1;

    wait_async0();                              // this chunk's DMA is in LDS
    __syncthreads();                            // ... for every wave
    if (c + 32 < SEQ) stage(c + 32);            // overlap next DMA with compute

    // ---- scores: 2 key tiles x 2 K-steps = 4 WMMAs, B-frags from LDS ----
    v8f sc[2];
#pragma unroll
    for (int g = 0; g < 2; ++g) {
      const unsigned short* kr = &kbuf[p][g * 16 + ln][0];   // B col N = key
      Frag b0, b1;
      const u32x4* p0 = (const u32x4*)(kr + half * 16);
      b0.u[0] = p0[0]; b0.u[1] = p0[1];
      const u32x4* p1 = (const u32x4*)(kr + 32 + half * 16);
      b1.u[0] = p1[0]; b1.u[1] = p1[1];
      v8f s = {};
      s = wmma_bf16(aq0, b0, s);
      s = wmma_bf16(aq1, b1, s);
      sc[g] = s;
    }

    // ---- online softmax (row = D-layout VGPR index; 16-lane row reductions) ----
#pragma unroll
    for (int r = 0; r < 8; ++r) {
      float mx = fmaxf(sc[0][r], sc[1][r]);
#pragma unroll
      for (int off = 8; off >= 1; off >>= 1)
        mx = fmaxf(mx, __shfl_xor(mx, off, 32));
      const float mn    = fmaxf(m[r], mx);
      const float alpha = __expf(m[r] - mn);
      const float p0 = __expf(sc[0][r] - mn);
      const float p1 = __expf(sc[1][r] - mn);
      float rs = p0 + p1;
#pragma unroll
      for (int off = 8; off >= 1; off >>= 1)
        rs += __shfl_xor(rs, off, 32);
      l[r] = l[r] * alpha + rs;
      m[r] = mn;
#pragma unroll
      for (int t = 0; t < 4; ++t) oacc[t][r] *= alpha;
      // stage P (bf16) row-major in wave-private LDS: row = M, col = key-in-chunk
      pbuf[wave][r + 8 * half][ln]      = f2bf(p0);
      pbuf[wave][r + 8 * half][16 + ln] = f2bf(p1);
    }

    // ---- reload P as a 16x32 A-fragment (HW keeps per-wave DS ops in order) ----
    Frag ap;
    {
      const unsigned short* prow = &pbuf[wave][ln][0];
      const u32x4* pp = (const u32x4*)(prow + half * 16);
      ap.u[0] = pp[0];     // K = half*16 + [0..7]
      ap.u[1] = pp[1];     // K = half*16 + 8 + [0..7]
    }

    // ---- O += P @ V : 4 WMMAs over Dh, B-frags from LDS ----
#pragma unroll
    for (int t = 0; t < 4; ++t) {
      const unsigned short* vr = &vbuf[p][t * 16 + ln][0];   // B col N = head dim
      Frag bv;
      const u32x4* vp = (const u32x4*)(vr + half * 16);
      bv.u[0] = vp[0]; bv.u[1] = vp[1];
      oacc[t] = wmma_bf16(ap, bv, oacc[t]);
    }
  }

  // ---- normalize and store [b, s, h*64 + d] bf16 ----
#pragma unroll
  for (int r = 0; r < 8; ++r) {
    const float inv = 1.0f / l[r];
    const int s = q0 + r + 8 * half;
#pragma unroll
    for (int t = 0; t < 4; ++t) {
      const int col = h * D_HEAD + t * 16 + ln;
      o[((size_t)b * SEQ + s) * D_MODEL + col] = f2bf(oacc[t][r] * inv);
    }
  }
}

// ---------------- kernel 4: output projection + bias (f32 out) ----------------
// Same 32x64-per-wave blocking as the QKV GEMM.
__global__ void mha_proj(const unsigned short* __restrict__ A,   // ctx bf16 [NROWS, D]
                         const unsigned short* __restrict__ W,   // Wp bf16 [out, in]
                         const float* __restrict__ bias,
                         float* __restrict__ out) {
  const int wave = threadIdx.x >> 5;
  const int lane = threadIdx.x & 31;
  const int half = lane >> 4;
  const int ln   = lane & 15;

  const int m0 = blockIdx.x * 32;
  const int n0 = blockIdx.y * 256 + wave * 64;

  v8f acc[2][4];
#pragma unroll
  for (int mt = 0; mt < 2; ++mt)
#pragma unroll
    for (int nt = 0; nt < 4; ++nt) acc[mt][nt] = {};

  const unsigned short* aptr[2];
  aptr[0] = A + (size_t)(m0 + ln) * D_MODEL;
  aptr[1] = A + (size_t)(m0 + 16 + ln) * D_MODEL;
  const unsigned short* bptr[4];
#pragma unroll
  for (int nt = 0; nt < 4; ++nt)
    bptr[nt] = W + (size_t)(n0 + nt * 16 + ln) * D_MODEL;

  for (int k0 = 0; k0 < D_MODEL; k0 += 32) {
    Frag a[2], b[4];
#pragma unroll
    for (int mt = 0; mt < 2; ++mt) {
      const u32x4* ap = (const u32x4*)(aptr[mt] + k0 + half * 16);
      a[mt].u[0] = ap[0];
      a[mt].u[1] = ap[1];
    }
#pragma unroll
    for (int nt = 0; nt < 4; ++nt) {
      const u32x4* bp = (const u32x4*)(bptr[nt] + k0 + half * 16);
      b[nt].u[0] = bp[0];
      b[nt].u[1] = bp[1];
    }
    if (k0 + 32 < D_MODEL) {
      __builtin_prefetch(aptr[0] + k0 + 32, 0, 3);
      __builtin_prefetch(aptr[1] + k0 + 32, 0, 3);
      __builtin_prefetch(bptr[0] + k0 + 32, 0, 3);
      __builtin_prefetch(bptr[2] + k0 + 32, 0, 3);
    }
#pragma unroll
    for (int nt = 0; nt < 4; ++nt)
#pragma unroll
      for (int mt = 0; mt < 2; ++mt)
        acc[mt][nt] = wmma_bf16(a[mt], b[nt], acc[mt][nt]);
  }

  float bv[4];
#pragma unroll
  for (int nt = 0; nt < 4; ++nt) bv[nt] = bias[n0 + nt * 16 + ln];

#pragma unroll
  for (int mt = 0; mt < 2; ++mt)
#pragma unroll
    for (int nt = 0; nt < 4; ++nt)
#pragma unroll
      for (int r = 0; r < 8; ++r) {
        const int row = m0 + mt * 16 + r + 8 * half;
        const int col = n0 + nt * 16 + ln;
        out[(size_t)row * D_MODEL + col] = acc[mt][nt][r] + bv[nt];
      }
}

// ---------------- host launcher ----------------
extern "C" void kernel_launch(void* const* d_in, const int* in_sizes, int n_in,
                              void* d_out, int out_size, void* d_ws, size_t ws_size,
                              hipStream_t stream) {
  const float* x  = (const float*)d_in[0];
  // d_in[1] = attn_weights (unused by reference forward)
  const float* Wq = (const float*)d_in[2];
  const float* Wk = (const float*)d_in[3];
  const float* Wv = (const float*)d_in[4];
  const float* Wp = (const float*)d_in[5];
  const float* bp = (const float*)d_in[6];
  float* out = (float*)d_out;

  unsigned short* ws = (unsigned short*)d_ws;
  size_t off = 0;
  unsigned short* x_bf  = ws + off; off += (size_t)NROWS * D_MODEL;    // 8 MB
  unsigned short* wq_bf = ws + off; off += (size_t)D_MODEL * D_MODEL;  // 2 MB
  unsigned short* wk_bf = ws + off; off += (size_t)D_MODEL * D_MODEL;
  unsigned short* wv_bf = ws + off; off += (size_t)D_MODEL * D_MODEL;
  unsigned short* wp_bf = ws + off; off += (size_t)D_MODEL * D_MODEL;
  unsigned short* q_bf  = ws + off; off += (size_t)NROWS * D_MODEL;    // [B,H,S,Dh]
  unsigned short* k_bf  = ws + off; off += (size_t)NROWS * D_MODEL;    // [B,H,S,Dh]
  unsigned short* vT_bf = ws + off; off += (size_t)NROWS * D_MODEL;    // [B,H,Dh,S]
  unsigned short* ctx_bf = ws + off; off += (size_t)NROWS * D_MODEL;   // [NROWS, D]
  // total ~50.3 MB of workspace (fits comfortably in the 192 MB L2)

  const int nx = NROWS * D_MODEL;
  const int nw = D_MODEL * D_MODEL;
  mha_cvt_bf16<<<nx / 8 / 256, 256, 0, stream>>>(x,  x_bf,  nx / 8);
  mha_cvt_bf16<<<nw / 8 / 256, 256, 0, stream>>>(Wq, wq_bf, nw / 8);
  mha_cvt_bf16<<<nw / 8 / 256, 256, 0, stream>>>(Wk, wk_bf, nw / 8);
  mha_cvt_bf16<<<nw / 8 / 256, 256, 0, stream>>>(Wv, wv_bf, nw / 8);
  mha_cvt_bf16<<<nw / 8 / 256, 256, 0, stream>>>(Wp, wp_bf, nw / 8);

  dim3 gemm_grid(NROWS / 32, D_MODEL / 256);    // (128, 4), 4 waves/block
  mha_gemm_qkv<<<gemm_grid, 128, 0, stream>>>(x_bf, wq_bf, q_bf, 0);
  mha_gemm_qkv<<<gemm_grid, 128, 0, stream>>>(x_bf, wk_bf, k_bf, 0);
  mha_gemm_qkv<<<gemm_grid, 128, 0, stream>>>(x_bf, wv_bf, vT_bf, 1);

  mha_attn<<<dim3(SEQ / 64, BATCH * NUM_HEADS), 128, 0, stream>>>(q_bf, k_bf, vT_bf, ctx_bf);

  mha_proj<<<gemm_grid, 128, 0, stream>>>(ctx_bf, wp_bf, bp, out);
}
